// GATLayer_38019050504254
// MI455X (gfx1250) — matrix operand order repaired and checked by reading, not compiled, not run on hardware
//
#include <hip/hip_runtime.h>
#include <hip/hip_bf16.h>
#include <cstdint>
#include <cstddef>

#define BB  8
#define NN  1024
#define DD  256
#define HDS 4
#define DH  64

typedef __attribute__((ext_vector_type(16))) __bf16 v16bf;
typedef __attribute__((ext_vector_type(8)))  float  v8f;
typedef __attribute__((ext_vector_type(4)))  int    i4v;

#if defined(__HIP_DEVICE_COMPILE__) && defined(__has_builtin)
#if __has_builtin(__builtin_amdgcn_global_load_async_to_lds_b128)
#define HAVE_ASYNC_LDS 1
#endif
#if __has_builtin(__builtin_amdgcn_s_wait_asynccnt)
#define HAVE_WAIT_ASYNC_BUILTIN 1
#endif
#endif

// addrspace-qualified int4 pointers for the async builtin
typedef __attribute__((address_space(1))) i4v gi4;   // global
typedef __attribute__((address_space(3))) i4v li4;   // LDS

__device__ __forceinline__ void async_cp_b128(const void* g, void* lds) {
#ifdef HAVE_ASYNC_LDS
  __builtin_amdgcn_global_load_async_to_lds_b128((gi4*)g, (li4*)lds, 0, 0);
#else
  *(uint4*)lds = *(const uint4*)g;   // fallback: sync copy through VGPRs
#endif
}

__device__ __forceinline__ void wait_async0() {
#ifdef HAVE_ASYNC_LDS
#ifdef HAVE_WAIT_ASYNC_BUILTIN
  __builtin_amdgcn_s_wait_asynccnt(0);
#else
  asm volatile("s_wait_asynccnt 0x0" ::: "memory");
#endif
#endif
}

// ---------------------------------------------------------------------------
// Kernel 1: hW = h @ W^T via bf16 WMMA (fp32 accumulate).
// Also writes WhT bf16 in (b, h, dcol, n) layout so the AV-GEMM B operand
// is contiguous along j.
// ---------------------------------------------------------------------------
__global__ __launch_bounds__(256) void k_gemm_hW(const float* __restrict__ h,
                                                 const float* __restrict__ W,
                                                 float* __restrict__ hW,
                                                 __bf16* __restrict__ WhT) {
  const int lane = threadIdx.x & 31;
  const int wave = threadIdx.x >> 5;
  const int wid  = blockIdx.x * 8 + wave;   // 2048 waves total
  const int rt   = wid >> 2;                // 512 row tiles (16 rows each)
  const int cg   = wid & 3;                 // 4 column groups of 64
  const int half = lane >> 4;
  const int r    = lane & 15;
  const int row  = rt * 16 + r;             // global row in flattened (B*N)

  v8f acc[4] = {};

  for (int kc = 0; kc < DD; kc += 32) {
    // A tile 16x32 of h: lane owns row r; K groups per ISA 16-bit A layout
    const float* ap = h + (size_t)row * DD + kc + half * 8;
    float4 a0 = *(const float4*)(ap);
    float4 a1 = *(const float4*)(ap + 4);
    float4 a2 = *(const float4*)(ap + 16);
    float4 a3 = *(const float4*)(ap + 20);
    float af[16] = {a0.x,a0.y,a0.z,a0.w, a1.x,a1.y,a1.z,a1.w,
                    a2.x,a2.y,a2.z,a2.w, a3.x,a3.y,a3.z,a3.w};
    v16bf va;
#pragma unroll
    for (int e = 0; e < 16; ++e) va[e] = (__bf16)af[e];

#pragma unroll
    for (int c = 0; c < 4; ++c) {
      // B tile 32x16 of W^T: B[k][n] = W[n][k]; K contiguous in W row
      const int n = cg * 64 + c * 16 + r;
      const float* bp = W + (size_t)n * DD + kc + half * 8;
      float4 b0 = *(const float4*)(bp);
      float4 b1 = *(const float4*)(bp + 4);
      float4 b2 = *(const float4*)(bp + 16);
      float4 b3 = *(const float4*)(bp + 20);
      float bf_[16] = {b0.x,b0.y,b0.z,b0.w, b1.x,b1.y,b1.z,b1.w,
                       b2.x,b2.y,b2.z,b2.w, b3.x,b3.y,b3.z,b3.w};
      v16bf vb;
#pragma unroll
      for (int e = 0; e < 16; ++e) vb[e] = (__bf16)bf_[e];

      acc[c] = __builtin_amdgcn_wmma_f32_16x16x32_bf16(
          false, va, false, vb, (short)0, acc[c], false, false);
    }
  }

#pragma unroll
  for (int c = 0; c < 4; ++c) {
    const int col = cg * 64 + c * 16 + r;
    const int hh  = col >> 6;
    const int dc  = col & 63;
#pragma unroll
    for (int v = 0; v < 8; ++v) {
      const int M    = v + half * 8;          // C/D layout: lanes16-31 -> M+8
      const int grow = rt * 16 + M;
      const float fv = acc[c][v];
      hW[(size_t)grow * DD + col] = fv;
      const int bb = grow >> 10;
      const int ii = grow & (NN - 1);
      WhT[(((size_t)bb * HDS + hh) * DH + dc) * NN + ii] = (__bf16)fv;
    }
  }
}

// ---------------------------------------------------------------------------
// Kernel 2: s[b,h,i] = Wh[b,i,h,:] . a_src ; t likewise with a_dst.
// ---------------------------------------------------------------------------
__global__ __launch_bounds__(256) void k_st(const float* __restrict__ hW,
                                            const float* __restrict__ a_src,
                                            const float* __restrict__ a_dst,
                                            float* __restrict__ s,
                                            float* __restrict__ t) {
  const int gid = blockIdx.x * blockDim.x + threadIdx.x;
  if (gid >= BB * HDS * NN) return;
  const int i  = gid & (NN - 1);
  const int hh = (gid >> 10) & (HDS - 1);
  const int b  = gid >> 12;
  const float* wp = hW + ((size_t)(b * NN + i)) * DD + hh * DH;
  float ss = 0.f, tt = 0.f;
#pragma unroll
  for (int dd = 0; dd < DH; dd += 4) {
    float4 w  = *(const float4*)(wp + dd);
    float4 as = *(const float4*)(a_src + dd);
    float4 ad = *(const float4*)(a_dst + dd);
    ss += w.x * as.x + w.y * as.y + w.z * as.z + w.w * as.w;
    tt += w.x * ad.x + w.y * ad.y + w.z * ad.z + w.w * ad.w;
  }
  s[(size_t)(b * HDS + hh) * NN + i] = ss;
  t[(size_t)(b * HDS + hh) * NN + i] = tt;
}

// ---------------------------------------------------------------------------
// Kernel 3: fused logits -> single-pass softmax -> alpha@Wh (bf16 WMMA)
//           -> BN(eval) -> ELU -> residual.
// Block = 8 waves sharing one (b, h); wave w owns i-tile (itg*8 + w).
// Per 32-j chunk the block stages the shared 32x64 bf16 WhT slab (+ t chunk)
// into double-buffered LDS via GLOBAL_LOAD_ASYNC_TO_LDS, overlapped with the
// exp/WMMA compute of the previous chunk.
// ---------------------------------------------------------------------------
#define SB_STRIDE 40  /* bf16 elems per dcol row: 80B, bank-conflict-free */

#define PSTEP(idx, tv, av)                                                \
  do {                                                                    \
    float _x = sv + (tv);                                                 \
    _x = _x >= 0.f ? _x : 0.2f * _x;                                      \
    float _e = _x + 0.5f * (av);                                          \
    float _p = (__builtin_fabsf(av) < 1e-6f) ? 0.f : __expf(_e);          \
    psum += _p;                                                           \
    pa[idx] = (__bf16)_p;                                                 \
  } while (0)

__global__ __launch_bounds__(256) void k_attn(const float* __restrict__ adj,
                                              const float* __restrict__ hW,
                                              const __bf16* __restrict__ WhT,
                                              const float* __restrict__ s,
                                              const float* __restrict__ t,
                                              const float* __restrict__ gamma,
                                              const float* __restrict__ beta,
                                              const float* __restrict__ rmean,
                                              const float* __restrict__ rvar,
                                              float* __restrict__ out) {
  __shared__ __align__(16) __bf16 sB[2][DH * SB_STRIDE];  // 2 x 5120 B
  __shared__ __align__(16) float  sT[2][32];              // 2 x 128 B

  const int tid  = threadIdx.x;
  const int lane = tid & 31;
  const int wave = tid >> 5;
  // block -> (b, itg, h); wave -> i-tile
  const int hh  = blockIdx.x & 3;
  const int itg = (blockIdx.x >> 2) & 7;
  const int b   = blockIdx.x >> 5;
  const int it  = itg * 8 + wave;
  const int half = lane >> 4;
  const int r    = lane & 15;
  const int i    = it * 16 + r;             // A-operand row owned by lane

  const float*  adjRow = adj + ((size_t)b * NN + i) * NN;
  const float*  tp     = t + (size_t)(b * HDS + hh) * NN;
  const float   sv     = s[(size_t)(b * HDS + hh) * NN + i];
  const __bf16* wtp    = WhT + ((size_t)b * HDS + hh) * ((size_t)DH * NN);

  // cooperative stage of one 32-j chunk: 64 dcol rows x 32 j (bf16) + t chunk
  const int dcol = tid >> 2;   // 0..63
  const int seg  = tid & 3;    // 4 x 8 bf16 segments per row
  auto stage = [&](int buf, int jb) {
    async_cp_b128(wtp + (size_t)dcol * NN + jb + seg * 8,
                  &sB[buf][dcol * SB_STRIDE + seg * 8]);
    if (tid < 8)
      async_cp_b128(tp + jb + tid * 4, &sT[buf][tid * 4]);
  };

  v8f  acc[4] = {};
  float psum  = 0.f;

  stage(0, 0);

  for (int k = 0; k < NN / 32; ++k) {
    const int jb  = k * 32;
    const int cur = k & 1;

    wait_async0();        // this wave's staged chunk is in LDS
    __syncthreads();      // everyone's portion visible; prev reads done

    if (jb + 32 < NN) stage(cur ^ 1, jb + 32);   // overlap next chunk

    // lane's K-groups per ISA 16-bit A layout
    const int jA = jb + half * 8;
    const int jBo = jb + 16 + half * 8;
    float4 a0 = *(const float4*)(adjRow + jA);
    float4 a1 = *(const float4*)(adjRow + jA + 4);
    float4 a2 = *(const float4*)(adjRow + jBo);
    float4 a3 = *(const float4*)(adjRow + jBo + 4);
    if (jb + 32 < NN)
      __builtin_prefetch(adjRow + jb + 32 + half * 8, 0, 3);

    float4 t0 = *(const float4*)&sT[cur][half * 8];
    float4 t1 = *(const float4*)&sT[cur][half * 8 + 4];
    float4 t2 = *(const float4*)&sT[cur][16 + half * 8];
    float4 t3 = *(const float4*)&sT[cur][16 + half * 8 + 4];

    v16bf pa;
    PSTEP(0,  t0.x, a0.x); PSTEP(1,  t0.y, a0.y);
    PSTEP(2,  t0.z, a0.z); PSTEP(3,  t0.w, a0.w);
    PSTEP(4,  t1.x, a1.x); PSTEP(5,  t1.y, a1.y);
    PSTEP(6,  t1.z, a1.z); PSTEP(7,  t1.w, a1.w);
    PSTEP(8,  t2.x, a2.x); PSTEP(9,  t2.y, a2.y);
    PSTEP(10, t2.z, a2.z); PSTEP(11, t2.w, a2.w);
    PSTEP(12, t3.x, a3.x); PSTEP(13, t3.y, a3.y);
    PSTEP(14, t3.z, a3.z); PSTEP(15, t3.w, a3.w);

#pragma unroll
    for (int c = 0; c < 4; ++c) {
      // B operand from LDS: lane owns col (c*16+r), K = half*8 + 0..7, 16 + half*8 + 0..7
      const uint4* bp =
          (const uint4*)&sB[cur][(c * 16 + r) * SB_STRIDE + half * 8];
      union { struct { uint4 a, b; } u; v16bf v; } bv;
      bv.u.a = bp[0];
      bv.u.b = bp[2];
      acc[c] = __builtin_amdgcn_wmma_f32_16x16x32_bf16(
          false, pa, false, bv.v, (short)0, acc[c], false, false);
    }

    __syncthreads();      // all reads of this buffer done before re-stage
  }

  // finish softmax denominator: lane r and lane r+16 hold disjoint j subsets
  psum += __shfl_xor(psum, 16, 32);
  const float inv = 1.f / psum;

#pragma unroll
  for (int c = 0; c < 4; ++c) {
    const int col = hh * DH + c * 16 + r;
    const float mn = rmean[col];
    const float sc = gamma[col] * __frsqrt_rn(rvar[col] + 1e-5f);
    const float bt = beta[col];
#pragma unroll
    for (int v = 0; v < 8; ++v) {
      const int M = v + half * 8;                 // C/D row of this VGPR
      const float rinv = __shfl(inv, M, 32);      // row M denom lives in lane M
      const float val  = acc[c][v] * rinv;        // softmax normalize
      const float bn   = (val - mn) * sc + bt;    // BatchNorm (eval)
      const float el   = bn > 0.f ? bn : (__expf(bn) - 1.f);  // ELU
      const size_t go  = ((size_t)(b * NN + it * 16 + M)) * DD + col;
      out[go] = el + hW[go];                      // residual with W(h)
    }
  }
}

// ---------------------------------------------------------------------------
extern "C" void kernel_launch(void* const* d_in, const int* in_sizes, int n_in,
                              void* d_out, int out_size, void* d_ws, size_t ws_size,
                              hipStream_t stream) {
  const float* h      = (const float*)d_in[0];
  const float* adj    = (const float*)d_in[1];
  const float* W      = (const float*)d_in[2];
  const float* a_src  = (const float*)d_in[3];
  const float* a_dst  = (const float*)d_in[4];
  const float* gamma  = (const float*)d_in[5];
  const float* beta   = (const float*)d_in[6];
  const float* rmean  = (const float*)d_in[7];
  const float* rvar   = (const float*)d_in[8];
  float* out = (float*)d_out;

  // workspace: hW fp32 (8 MB) | WhT bf16 (4 MB) | s (128 KB) | t (128 KB)
  float*  hWf = (float*)d_ws;
  __bf16* WhT = (__bf16*)((char*)d_ws + (size_t)8 * 1024 * 1024);
  float*  sP  = (float*)((char*)d_ws + (size_t)12 * 1024 * 1024);
  float*  tP  = sP + BB * HDS * NN;

  k_gemm_hW<<<256, 256, 0, stream>>>(h, W, hWf, WhT);
  k_st<<<128, 256, 0, stream>>>(hWf, a_src, a_dst, sP, tP);
  k_attn<<<256, 256, 0, stream>>>(adj, hWf, WhT, sP, tP,
                                  gamma, beta, rmean, rvar, out);
}
